// GConvGRUNet_27573690040587
// MI455X (gfx1250) — matrix operand order repaired and checked by reading, not compiled, not run on hardware
//
#include <hip/hip_runtime.h>
#include <hip/hip_bf16.h>
#include <math.h>

typedef _Float16 v16h __attribute__((ext_vector_type(16)));
typedef _Float16 v8h  __attribute__((ext_vector_type(8)));
typedef float    v8f  __attribute__((ext_vector_type(8)));
typedef float    v4f  __attribute__((ext_vector_type(4)));

#define N_NODES   100000
#define CH        256
#define MTILE     32      // rows per wave (2 x 16-row WMMA tiles; 100000 % 32 == 0)
#define WAVES     8
#define LDS_STRIDE 264    // f16 elems per LDS row (256 + 8 pad; 264*2=528=33*16 keeps 16B align)

// ---------------------------------------------------------------------------
// Prep: transpose + f32->f16 convert the three live weight matrices.
// Wt[n*CH + k] = (f16) W[k*CH + n]  (N-major => contiguous-K per lane for the
// WMMA B-fragment).
// ---------------------------------------------------------------------------
__global__ void prep_weights(const float* __restrict__ Wxz,
                             const float* __restrict__ Wxh,
                             const float* __restrict__ Wlin,
                             _Float16* __restrict__ ws16) {
  const int n = blockIdx.x;
  const int k = threadIdx.x;
  const float* W = (blockIdx.y == 0) ? Wxz : ((blockIdx.y == 1) ? Wxh : Wlin);
  _Float16* Wt = ws16 + (size_t)blockIdx.y * CH * CH;
  Wt[n * CH + k] = (_Float16)W[k * CH + n];
}

__global__ void prep_bias(const float* __restrict__ bxz, const float* __restrict__ bhz,
                          const float* __restrict__ bxh, const float* __restrict__ bhh,
                          float* __restrict__ bz, float* __restrict__ bh) {
  const int t = threadIdx.x;
  bz[t] = bxz[t] + bhz[t];
  bh[t] = bxh[t] + bhh[t];
}

// ---------------------------------------------------------------------------
// Fused: Z/T GEMMs -> GRU elementwise -> ELU -> readout GEMM.
// One wave per 32-row slab (2 WMMA row-tiles share every B fragment);
// 8 waves per workgroup.
// ---------------------------------------------------------------------------
__global__ __launch_bounds__(256)
void gconv_gru_fused(const float*    __restrict__ x,
                     const _Float16* __restrict__ Wzt,
                     const _Float16* __restrict__ Wht,
                     const _Float16* __restrict__ Wlt,
                     const float*    __restrict__ bz,
                     const float*    __restrict__ bh,
                     const float*    __restrict__ blin,
                     float*          __restrict__ out) {
  __shared__ _Float16 hbuf[WAVES * MTILE * LDS_STRIDE];   // 132 KB

  const unsigned tid    = threadIdx.x;
  const unsigned wave   = tid >> 5;
  const unsigned lane   = tid & 31;
  const unsigned laneLo = lane & 15;
  const unsigned laneHi = lane >> 4;

  // Warm L2 with the (tiny, L2-resident) converted weights.
  __builtin_prefetch(Wzt + tid * 256, 0, 0);
  __builtin_prefetch(Wht + tid * 256, 0, 0);
  __builtin_prefetch(Wlt + tid * 256, 0, 0);

  const unsigned numRowBlocks = N_NODES / MTILE;                  // 3125 exact
  const unsigned rowBlock = blockIdx.x * WAVES + wave;
  const unsigned rowBase  = (rowBlock < numRowBlocks) ? rowBlock * MTILE : N_NODES;

  // ---- Load 32x256 f32 slab of X once; build 2x8 f16 A-fragments.
  // A 16x32 f16 (contiguous K, as B): lanes 0-15 hold row=laneLo, K 0..15;
  // lanes 16-31 hold row=laneLo, K 16..31.
  v16h afrag[2][8];
  #pragma unroll
  for (int rt = 0; rt < 2; ++rt) {
    unsigned aRow = rowBase + rt * 16 + laneLo;
    if (aRow >= N_NODES) aRow = N_NODES - 1;      // clamp: EXEC stays all-1s
    #pragma unroll
    for (int ks = 0; ks < 8; ++ks) {
      const float* p = x + (size_t)aRow * CH + ks * 32 + laneHi * 16;
      v4f f0 = *(const v4f*)(p);
      v4f f1 = *(const v4f*)(p + 4);
      v4f f2 = *(const v4f*)(p + 8);
      v4f f3 = *(const v4f*)(p + 12);
      v16h a;
      #pragma unroll
      for (int i = 0; i < 4; ++i) {
        a[i]      = (_Float16)f0[i];
        a[4 + i]  = (_Float16)f1[i];
        a[8 + i]  = (_Float16)f2[i];
        a[12 + i] = (_Float16)f3[i];
      }
      afrag[rt][ks] = a;
    }
  }

  _Float16* hb = hbuf + wave * MTILE * LDS_STRIDE;

  // ---- GEMM1 (Z) + GEMM2 (T) + GRU elementwise, h -> LDS (f16).
  for (int j = 0; j < 16; ++j) {
    const unsigned col = j * 16 + laneLo;
    v8f accZ0 = {}, accZ1 = {}, accT0 = {}, accT1 = {};
    #pragma unroll
    for (int ks = 0; ks < 8; ++ks) {
      // B 32x16 f16 (contiguous K, as A): lane = N column; lanes 0-15 K 0..15, lanes 16-31 K 16..31.
      v16h bfz = *(const v16h*)(Wzt + (size_t)col * CH + ks * 32 + laneHi * 16);
      v16h bfh = *(const v16h*)(Wht + (size_t)col * CH + ks * 32 + laneHi * 16);
      accZ0 = __builtin_amdgcn_wmma_f32_16x16x32_f16(false, afrag[0][ks], false, bfz,
                                                     (short)0, accZ0, false, false);
      accZ1 = __builtin_amdgcn_wmma_f32_16x16x32_f16(false, afrag[1][ks], false, bfz,
                                                     (short)0, accZ1, false, false);
      accT0 = __builtin_amdgcn_wmma_f32_16x16x32_f16(false, afrag[0][ks], false, bfh,
                                                     (short)0, accT0, false, false);
      accT1 = __builtin_amdgcn_wmma_f32_16x16x32_f16(false, afrag[1][ks], false, bfh,
                                                     (short)0, accT1, false, false);
    }
    const float bzv = bz[col];
    const float bhv = bh[col];
    // C layout: VGPR i -> row (i + 8*laneHi) of the tile, col = laneLo.
    #pragma unroll
    for (int rt = 0; rt < 2; ++rt) {
      #pragma unroll
      for (int i = 0; i < 8; ++i) {
        const float zp = (rt ? accZ1[i] : accZ0[i]) + bzv;
        const float tp = (rt ? accT1[i] : accT0[i]) + bhv;
        const float zg = 1.0f / (1.0f + expf(-zp));
        const float tg = tanhf(tp);
        const float Hv = (1.0f - zg) * tg;                 // H0 = 0
        const float hv = (Hv > 0.0f) ? Hv : expm1f(Hv);    // ELU(alpha=1)
        hb[(rt * 16 + i + laneHi * 8) * LDS_STRIDE + col] = (_Float16)hv;
      }
    }
  }

  __syncthreads();

  // ---- Reload h as A-fragments for the readout GEMM.
  v16h hfrag[2][8];
  #pragma unroll
  for (int rt = 0; rt < 2; ++rt) {
    #pragma unroll
    for (int ks = 0; ks < 8; ++ks) {
      const _Float16* hp = hb + (rt * 16 + laneLo) * LDS_STRIDE + ks * 32 + laneHi * 16;
      v8h h0 = *(const v8h*)(hp);        // K = ks*32 + 16*laneHi + 0..7
      v8h h1 = *(const v8h*)(hp + 8);    // K = ks*32 + 16*laneHi + 8..15
      v16h a;
      #pragma unroll
      for (int i = 0; i < 8; ++i) { a[i] = h0[i]; a[8 + i] = h1[i]; }
      hfrag[rt][ks] = a;
    }
  }

  // ---- GEMM3: out = h @ W_lin + b_lin
  for (int j = 0; j < 16; ++j) {
    const unsigned col = j * 16 + laneLo;
    v8f acc0 = {}, acc1 = {};
    #pragma unroll
    for (int ks = 0; ks < 8; ++ks) {
      v16h bf = *(const v16h*)(Wlt + (size_t)col * CH + ks * 32 + laneHi * 16);
      acc0 = __builtin_amdgcn_wmma_f32_16x16x32_f16(false, hfrag[0][ks], false, bf,
                                                    (short)0, acc0, false, false);
      acc1 = __builtin_amdgcn_wmma_f32_16x16x32_f16(false, hfrag[1][ks], false, bf,
                                                    (short)0, acc1, false, false);
    }
    const float bl = blin[col];
    #pragma unroll
    for (int rt = 0; rt < 2; ++rt) {
      #pragma unroll
      for (int i = 0; i < 8; ++i) {
        const unsigned r = rowBase + rt * 16 + i + laneHi * 8;
        if (r < N_NODES) out[(size_t)r * CH + col] = (rt ? acc1[i] : acc0[i]) + bl;
      }
    }
  }
}

// ---------------------------------------------------------------------------
extern "C" void kernel_launch(void* const* d_in, const int* in_sizes, int n_in,
                              void* d_out, int out_size, void* d_ws, size_t ws_size,
                              hipStream_t stream) {
  (void)in_sizes; (void)n_in; (void)out_size; (void)ws_size;
  // setup_inputs order:
  // 0:x 1:edge_index 2:edge_weight 3:W_xz 4:b_xz 5:W_hz 6:b_hz 7:W_xr 8:b_xr
  // 9:W_hr 10:b_hr 11:W_xh 12:b_xh 13:W_hh 14:b_hh 15:W_lin 16:b_lin
  const float* x    = (const float*)d_in[0];
  const float* Wxz  = (const float*)d_in[3];
  const float* bxz  = (const float*)d_in[4];
  const float* bhz  = (const float*)d_in[6];
  const float* Wxh  = (const float*)d_in[11];
  const float* bxh  = (const float*)d_in[12];
  const float* bhh  = (const float*)d_in[14];
  const float* Wlin = (const float*)d_in[15];
  const float* blin = (const float*)d_in[16];

  // Workspace layout: [Wzt | Wht | Wlt] f16 (3*128KB), then bz, bh (f32).
  _Float16* ws16 = (_Float16*)d_ws;
  float* bz = (float*)((char*)d_ws + (size_t)3 * CH * CH * sizeof(_Float16));
  float* bh = bz + CH;

  prep_weights<<<dim3(CH, 3), CH, 0, stream>>>(Wxz, Wxh, Wlin, ws16);
  prep_bias<<<1, CH, 0, stream>>>(bxz, bhz, bxh, bhh, bz, bh);

  const _Float16* Wzt = ws16;
  const _Float16* Wht = ws16 + CH * CH;
  const _Float16* Wlt = ws16 + 2 * CH * CH;

  const unsigned numRowBlocks = N_NODES / MTILE;                 // 3125
  const unsigned blocks = (numRowBlocks + WAVES - 1) / WAVES;    // 391
  gconv_gru_fused<<<blocks, 256, 0, stream>>>(x, Wzt, Wht, Wlt, bz, bh, blin,
                                              (float*)d_out);
}
